// MultiScaleRetention_49770081026345
// MI455X (gfx1250) — compile-verified, matrix operand order checked by reading, not run on hardware
//
#include <hip/hip_runtime.h>
#include <hip/hip_bf16.h>
#include <stdint.h>

// ---------------------------------------------------------------------------
// MultiScaleRetention (RetNet) for MI455X / gfx1250, wave32 + WMMA bf16.
// GEMM tiles staged to LDS with gfx1250 async-to-LDS copies (ASYNCcnt).
// ---------------------------------------------------------------------------

typedef __bf16 bf16_t;
typedef __attribute__((ext_vector_type(16))) __bf16 v16bf;
typedef __attribute__((ext_vector_type(8)))  __bf16 v8bf;
typedef __attribute__((ext_vector_type(8)))  float  v8f;

// Problem constants
constexpr int kB  = 2;
constexpr int kT  = 4096;
constexpr int kE  = 2048;
constexpr int kH  = 16;
constexpr int kKD = 128;   // E / H
constexpr int kHD = 256;   // E*FACTOR / H
constexpr int kC  = 256;   // CHUNK
constexpr int kN  = 16;    // T / CHUNK
constexpr int kE2 = 4096;  // E * FACTOR
constexpr float kEPS = 1e-5f;

// ---------------------------------------------------------------------------
// helpers
// ---------------------------------------------------------------------------
__device__ inline v8f vzero8() {
  v8f z;
#pragma unroll
  for (int i = 0; i < 8; i++) z[i] = 0.f;
  return z;
}

// Load one 16-lane-striped bf16 WMMA fragment: per lane, 8 contiguous bf16 at p
// plus 8 contiguous bf16 at p+16 (ISA 16-bit A/B layout: K groups of 8, +16).
__device__ inline v16bf load_frag(const bf16_t* p) {
  v8bf lo = *(const v8bf*)(p);
  v8bf hi = *(const v8bf*)(p + 16);
  return __builtin_shufflevector(lo, hi, 0, 1, 2, 3, 4, 5, 6, 7,
                                 8, 9, 10, 11, 12, 13, 14, 15);
}

__device__ inline v8f wmma_bf16(v16bf a, v16bf b, v8f c) {
  return __builtin_amdgcn_wmma_f32_16x16x32_bf16(
      /*neg_a=*/false, a, /*neg_b=*/false, b,
      /*c_mod=*/(short)0, c, /*reuse_a=*/false, /*reuse_b=*/false);
}

// gfx1250 async copy: 32 bytes global -> LDS per lane (two B128 async ops).
// INST_OFFSET is added to BOTH the LDS and the global address (ISA 15.18.3),
// so one address pair covers both halves. Tracked by ASYNCcnt.
__device__ inline void async_copy32(bf16_t* lds_dst, const bf16_t* gsrc) {
  unsigned lds_off = (unsigned)(uintptr_t)lds_dst;            // low 32 bits = LDS offset
  unsigned long long gaddr = (unsigned long long)(uintptr_t)gsrc;
  asm volatile("global_load_async_to_lds_b128 %0, %1, off"
               :: "v"(lds_off), "v"(gaddr) : "memory");
  asm volatile("global_load_async_to_lds_b128 %0, %1, off offset:16"
               :: "v"(lds_off), "v"(gaddr) : "memory");
}

__device__ inline void async_wait_all() {
  asm volatile("s_wait_asynccnt 0" ::: "memory");
}

// ---------------------------------------------------------------------------
// K0: f32 -> bf16 conversion
// ---------------------------------------------------------------------------
__global__ void msr_f32_to_bf16(const float* __restrict__ src,
                                bf16_t* __restrict__ dst, int n) {
  int i = blockIdx.x * blockDim.x + threadIdx.x;
  if (i < n) dst[i] = (bf16_t)src[i];
}

// ---------------------------------------------------------------------------
// K1/K8: generic bf16 WMMA GEMM:  C[M,N] = (A[M,K] * W[N,K]^T + bias) * alpha
// Block tile 128x128, async-LDS-staged 128x32 operand tiles, 8 waves (4x2),
// each wave owns 2x4 WMMA tiles.
// ---------------------------------------------------------------------------
template <bool BF16OUT>
__global__ void __launch_bounds__(256)
msr_gemm_bf16(const bf16_t* __restrict__ A, const bf16_t* __restrict__ W,
              const float* __restrict__ bias, float alpha,
              int M, int N, int K,
              float* __restrict__ Cf, bf16_t* __restrict__ Cb) {
  __shared__ bf16_t sA[128 * 32];
  __shared__ bf16_t sW[128 * 32];
  const int tid  = threadIdx.x;
  const int lane = tid & 31;
  const int wave = tid >> 5;
  const int wr   = wave >> 1;   // 0..3  (m direction)
  const int wc   = wave & 1;    // 0..1  (n direction)
  const int hf   = lane >> 4;
  const int ln   = lane & 15;
  const int bm   = blockIdx.y * 128;
  const int bn   = blockIdx.x * 128;

  v8f acc[2][4];
#pragma unroll
  for (int i = 0; i < 2; i++)
#pragma unroll
    for (int j = 0; j < 4; j++) acc[i][j] = vzero8();

  const int row_ld = tid >> 1;         // 0..127
  const int col_ld = (tid & 1) * 16;   // 0 or 16

  const bf16_t* ga = A + (size_t)(bm + row_ld) * K + col_ld;
  const bf16_t* gw = W + (size_t)(bn + row_ld) * K + col_ld;
  bf16_t* la = sA + row_ld * 32 + col_ld;
  bf16_t* lw = sW + row_ld * 32 + col_ld;

  for (int k0 = 0; k0 < K; k0 += 32) {
    // async global -> LDS stage (no VGPR round trip; ASYNCcnt tracked)
    async_copy32(la, ga + k0);
    async_copy32(lw, gw + k0);
    if (k0 + 32 < K) {                       // warm L2 for the next tile
      __builtin_prefetch(ga + k0 + 32, 0, 1);
      __builtin_prefetch(gw + k0 + 32, 0, 1);
    }
    async_wait_all();
    __syncthreads();

    v16bf af[2], wf[4];
#pragma unroll
    for (int i = 0; i < 2; i++)
      af[i] = load_frag(sA + (wr * 32 + i * 16 + ln) * 32 + (hf << 3));
#pragma unroll
    for (int j = 0; j < 4; j++)
      wf[j] = load_frag(sW + (wc * 64 + j * 16 + ln) * 32 + (hf << 3));
#pragma unroll
    for (int i = 0; i < 2; i++)
#pragma unroll
      for (int j = 0; j < 4; j++) acc[i][j] = wmma_bf16(af[i], wf[j], acc[i][j]);
    __syncthreads();
  }

#pragma unroll
  for (int i = 0; i < 2; i++) {
#pragma unroll
    for (int j = 0; j < 4; j++) {
      const int col = bn + wc * 64 + j * 16 + ln;
      const float bv = bias[col];
#pragma unroll
      for (int e = 0; e < 8; e++) {
        const int row = bm + wr * 32 + i * 16 + e + 8 * hf;
        const float v = (acc[i][j][e] + bv) * alpha;
        if (BF16OUT) Cb[(size_t)row * N + col] = (bf16_t)v;
        else         Cf[(size_t)row * N + col] = v;
      }
    }
  }
}

// ---------------------------------------------------------------------------
// K2: rotary (theta_shift) on q,k; emits head-major bf16 buffers:
//   qrb,qdb,krb: [B,H,T,KD]  ;  krtb: [B,H,N,KD,C] (chunk-transposed kr)
// ---------------------------------------------------------------------------
__global__ void msr_rotary(const bf16_t* __restrict__ qb,
                           const bf16_t* __restrict__ kb,
                           const float* __restrict__ sinp,
                           const float* __restrict__ cosp,
                           const float* __restrict__ inner_decay,
                           bf16_t* __restrict__ qrb, bf16_t* __restrict__ qdb,
                           bf16_t* __restrict__ krb, bf16_t* __restrict__ krtb) {
  int idx = blockIdx.x * blockDim.x + threadIdx.x;  // over B*T*E/2
  if (idx >= kB * kT * kE / 2) return;
  const int p  = idx % (kKD / 2);
  const int h  = (idx / (kKD / 2)) % kH;
  const int t  = (idx / (kE / 2)) % kT;
  const int b  = idx / (kT * (kE / 2));
  const int kd = p * 2;

  const float s0 = sinp[t * kKD + kd],     s1 = sinp[t * kKD + kd + 1];
  const float c0 = cosp[t * kKD + kd],     c1 = cosp[t * kKD + kd + 1];

  const size_t src = ((size_t)b * kT + t) * kE + h * kKD + kd;
  const float q0 = (float)qb[src], q1 = (float)qb[src + 1];
  const float k0 = (float)kb[src], k1 = (float)kb[src + 1];

  // rotate_every_two: even slot gets -x_odd, odd slot gets x_even
  const float qr0 = q0 * c0 - q1 * s0;
  const float qr1 = q1 * c1 + q0 * s1;
  const float kr0 = k0 * c0 - k1 * s0;
  const float kr1 = k1 * c1 + k0 * s1;

  const int c = t & (kC - 1);
  const int n = t >> 8;
  const float idec = inner_decay[h * kC + c];

  const size_t dst = (((size_t)(b * kH + h)) * kT + t) * kKD + kd;
  qrb[dst]     = (bf16_t)qr0;          qrb[dst + 1] = (bf16_t)qr1;
  qdb[dst]     = (bf16_t)(qr0 * idec); qdb[dst + 1] = (bf16_t)(qr1 * idec);
  krb[dst]     = (bf16_t)kr0;          krb[dst + 1] = (bf16_t)kr1;

  const size_t tdst =
      ((((size_t)(b * kH + h)) * kN + n) * kKD + kd) * kC + c;
  krtb[tdst]      = (bf16_t)kr0;
  krtb[tdst + kC] = (bf16_t)kr1;
}

// ---------------------------------------------------------------------------
// K4: pack v -> vtb [B,N,H,HD,C] and vmtb = vtb * mask[h, C-1, c]
// ---------------------------------------------------------------------------
__global__ void msr_pack_v(const bf16_t* __restrict__ vb,
                           const float* __restrict__ mask,
                           bf16_t* __restrict__ vtb, bf16_t* __restrict__ vmtb) {
  int idx = blockIdx.x * blockDim.x + threadIdx.x;  // over B*T*E2
  if (idx >= kB * kT * kE2) return;
  const int hd = idx % kHD;
  const int h  = (idx / kHD) % kH;
  const int t  = (idx / kE2) % kT;
  const int b  = idx / (kT * kE2);
  const bf16_t val = vb[idx];
  const int c = t & (kC - 1);
  const int n = t >> 8;
  const size_t d =
      (((size_t)((b * kN + n) * kH + h)) * kHD + hd) * kC + c;
  vtb[d]  = val;
  vmtb[d] = (bf16_t)((float)val * mask[((size_t)h * kC + (kC - 1)) * kC + c]);
}

// ---------------------------------------------------------------------------
// K3: per-(b,n,h) qk = (qr @ kr^T) * mask ; inner_scale = clip(|qk|.sum(-1),1)
//     qkb = bf16(qk / inner_scale). One block = full 256x256 tile, 8 waves,
//     each wave owns 2 tile-rows (row sums stay wave-local -> shfl reduce).
// ---------------------------------------------------------------------------
__global__ void __launch_bounds__(256)
msr_qk(const bf16_t* __restrict__ qrb, const bf16_t* __restrict__ krb,
       const float* __restrict__ mask,
       bf16_t* __restrict__ qkb, float* __restrict__ iscale) {
  const int bnh = blockIdx.x;
  const int h = bnh % kH;
  const int n = (bnh / kH) % kN;
  const int b = bnh / (kH * kN);
  const int lane = threadIdx.x & 31, wave = threadIdx.x >> 5;
  const int hf = lane >> 4, ln = lane & 15;

  const bf16_t* qch = qrb + (((size_t)(b * kH + h)) * kT + n * kC) * kKD;
  const bf16_t* kch = krb + (((size_t)(b * kH + h)) * kT + n * kC) * kKD;
  const size_t qkbase = (size_t)bnh * kC * kC;

#pragma unroll
  for (int trr = 0; trr < 2; trr++) {
    const int tr = wave * 2 + trr;
    v16bf afr[4];
#pragma unroll
    for (int kk = 0; kk < 4; kk++)
      afr[kk] = load_frag(qch + (size_t)(tr * 16 + ln) * kKD + kk * 32 + (hf << 3));

    v8f acc[16];
#pragma unroll
    for (int tc = 0; tc < 16; tc++) {
      v8f a = vzero8();
#pragma unroll
      for (int kk = 0; kk < 4; kk++) {
        v16bf bfr =
            load_frag(kch + (size_t)(tc * 16 + ln) * kKD + kk * 32 + (hf << 3));
        a = wmma_bf16(afr[kk], bfr, a);
      }
      acc[tc] = a;
    }

    float rs[8];
#pragma unroll
    for (int e = 0; e < 8; e++) rs[e] = 0.f;
#pragma unroll
    for (int tc = 0; tc < 16; tc++) {
#pragma unroll
      for (int e = 0; e < 8; e++) {
        const int row = tr * 16 + e + 8 * hf;
        const int col = tc * 16 + ln;
        const float v = acc[tc][e] * mask[((size_t)h * kC + row) * kC + col];
        acc[tc][e] = v;
        rs[e] += fabsf(v);
      }
    }
#pragma unroll
    for (int e = 0; e < 8; e++) {
      rs[e] += __shfl_xor(rs[e], 1);
      rs[e] += __shfl_xor(rs[e], 2);
      rs[e] += __shfl_xor(rs[e], 4);
      rs[e] += __shfl_xor(rs[e], 8);
      rs[e] = fmaxf(rs[e], 1.0f);
    }
    if (ln == 0) {
#pragma unroll
      for (int e = 0; e < 8; e++)
        iscale[(size_t)bnh * kC + tr * 16 + e + 8 * hf] = rs[e];
    }
#pragma unroll
    for (int tc = 0; tc < 16; tc++) {
#pragma unroll
      for (int e = 0; e < 8; e++) {
        const int row = tr * 16 + e + 8 * hf;
        const int col = tc * 16 + ln;
        qkb[qkbase + (size_t)row * kC + col] = (bf16_t)(acc[tc][e] / rs[e]);
      }
    }
  }
}

// ---------------------------------------------------------------------------
// K5: inner_output[b,n,h] = qk(256x256) @ v(256x256)   (f32 out)
// ---------------------------------------------------------------------------
__global__ void __launch_bounds__(256)
msr_inner(const bf16_t* __restrict__ qkb, const bf16_t* __restrict__ vtb,
          float* __restrict__ innerf) {
  const int bnh = blockIdx.x;
  const int lane = threadIdx.x & 31, wave = threadIdx.x >> 5;
  const int hf = lane >> 4, ln = lane & 15;
  const bf16_t* aB = qkb + (size_t)bnh * kC * kC;
  const bf16_t* bB = vtb + (size_t)bnh * kHD * kC;
  float* out = innerf + (size_t)bnh * kC * kHD;

#pragma unroll
  for (int trr = 0; trr < 2; trr++) {
    const int tr = wave * 2 + trr;
    v8f acc[16];
#pragma unroll
    for (int tc = 0; tc < 16; tc++) acc[tc] = vzero8();
    for (int kk = 0; kk < 8; kk++) {
      v16bf af = load_frag(aB + (size_t)(tr * 16 + ln) * kC + kk * 32 + (hf << 3));
#pragma unroll
      for (int tc = 0; tc < 16; tc++) {
        v16bf bfr =
            load_frag(bB + (size_t)(tc * 16 + ln) * kC + kk * 32 + (hf << 3));
        acc[tc] = wmma_bf16(af, bfr, acc[tc]);
      }
    }
#pragma unroll
    for (int tc = 0; tc < 16; tc++)
#pragma unroll
      for (int e = 0; e < 8; e++)
        out[(size_t)(tr * 16 + e + 8 * hf) * kHD + tc * 16 + ln] = acc[tc][e];
  }
}

// ---------------------------------------------------------------------------
// K5b: kv[b,n,h] = kr^T(128x256) @ (v*mask_last)(256x256)  (f32 out)
// ---------------------------------------------------------------------------
__global__ void __launch_bounds__(256)
msr_kv(const bf16_t* __restrict__ krtb, const bf16_t* __restrict__ vmtb,
       float* __restrict__ kvf) {
  const int bnh = blockIdx.x;
  const int h = bnh % kH;
  const int n = (bnh / kH) % kN;
  const int b = bnh / (kH * kN);
  const int lane = threadIdx.x & 31, wave = threadIdx.x >> 5;
  const int hf = lane >> 4, ln = lane & 15;

  const bf16_t* aB =
      krtb + (((size_t)(b * kH + h)) * kN + n) * (size_t)kKD * kC;
  const bf16_t* bB = vmtb + (size_t)bnh * kHD * kC;
  float* out = kvf + (size_t)bnh * kKD * kHD;

  const int tr = wave;  // 8 waves -> 8 kd tile-rows
  v8f acc[16];
#pragma unroll
  for (int tc = 0; tc < 16; tc++) acc[tc] = vzero8();
  for (int kk = 0; kk < 8; kk++) {
    v16bf af = load_frag(aB + (size_t)(tr * 16 + ln) * kC + kk * 32 + (hf << 3));
#pragma unroll
    for (int tc = 0; tc < 16; tc++) {
      v16bf bfr =
          load_frag(bB + (size_t)(tc * 16 + ln) * kC + kk * 32 + (hf << 3));
      acc[tc] = wmma_bf16(af, bfr, acc[tc]);
    }
  }
#pragma unroll
  for (int tc = 0; tc < 16; tc++)
#pragma unroll
    for (int e = 0; e < 8; e++)
      out[(size_t)(tr * 16 + e + 8 * hf) * kHD + tc * 16 + ln] = acc[tc][e];
}

// ---------------------------------------------------------------------------
// K6: sequential KV scan over chunks. 32 blocks = (b,h); thread = hd column;
// 128-deep f32 state in registers. Emits kv_rec^T bf16 [B,N,H,HD,KD] + scale.
// ---------------------------------------------------------------------------
__global__ void __launch_bounds__(256)
msr_scan(const float* __restrict__ kvf, const float* __restrict__ cross_decay,
         bf16_t* __restrict__ kvrect, float* __restrict__ cscale) {
  const int bh = blockIdx.x;
  const int h = bh % kH;
  const int b = bh / kH;
  const int t = threadIdx.x;  // hd
  const float cd = cross_decay[h];

  float st[kKD];
#pragma unroll
  for (int i = 0; i < kKD; i++) st[i] = 0.f;
  float scale = 1.f;

  for (int n = 0; n < kN; n++) {
    const int bnh = (b * kN + n) * kH + h;
    bf16_t* rec = kvrect + ((size_t)bnh * kHD + t) * kKD;
    const float inv = 1.f / scale;
#pragma unroll
    for (int i = 0; i < kKD; i++) rec[i] = (bf16_t)(st[i] * inv);
    cscale[(size_t)bnh * kHD + t] = scale;

    const float* kvi = kvf + (size_t)bnh * kKD * kHD;
    float s = 0.f;
#pragma unroll
    for (int i = 0; i < kKD; i++) {
      st[i] = st[i] * cd + kvi[(size_t)i * kHD + t];
      s += fabsf(st[i]);
    }
    scale = fmaxf(s, 1.f);
  }
}

// ---------------------------------------------------------------------------
// K7: cross_output = (qr*inner_decay) @ kv_rec ; combine, groupnorm over HD,
//     gate with silu(g) -> bf16 [B,T,E2]
// ---------------------------------------------------------------------------
__global__ void __launch_bounds__(256)
msr_cross(const bf16_t* __restrict__ qdb, const bf16_t* __restrict__ kvrect,
          const float* __restrict__ innerf, const float* __restrict__ cscale,
          const float* __restrict__ iscale, const bf16_t* __restrict__ gb,
          bf16_t* __restrict__ outg) {
  const int bnh = blockIdx.x;
  const int h = bnh % kH;
  const int n = (bnh / kH) % kN;
  const int b = bnh / (kH * kN);
  const int lane = threadIdx.x & 31, wave = threadIdx.x >> 5;
  const int hf = lane >> 4, ln = lane & 15;

  const bf16_t* aB = qdb + (((size_t)(b * kH + h)) * kT + n * kC) * kKD;
  const bf16_t* bB = kvrect + (size_t)bnh * kHD * kKD;
  const float* inn = innerf + (size_t)bnh * kC * kHD;

#pragma unroll
  for (int trr = 0; trr < 2; trr++) {
    const int tr = wave * 2 + trr;
    v8f acc[16];
#pragma unroll
    for (int tc = 0; tc < 16; tc++) acc[tc] = vzero8();
#pragma unroll
    for (int kk = 0; kk < 4; kk++) {
      v16bf af = load_frag(aB + (size_t)(tr * 16 + ln) * kKD + kk * 32 + (hf << 3));
#pragma unroll
      for (int tc = 0; tc < 16; tc++) {
        v16bf bfr =
            load_frag(bB + (size_t)(tc * 16 + ln) * kKD + kk * 32 + (hf << 3));
        acc[tc] = wmma_bf16(af, bfr, acc[tc]);
      }
    }

    float isc[8];
#pragma unroll
    for (int e = 0; e < 8; e++)
      isc[e] = iscale[(size_t)bnh * kC + tr * 16 + e + 8 * hf];

    float rsum[8], rsq[8];
#pragma unroll
    for (int e = 0; e < 8; e++) { rsum[e] = 0.f; rsq[e] = 0.f; }

#pragma unroll
    for (int tc = 0; tc < 16; tc++) {
      const float csc = cscale[(size_t)bnh * kHD + tc * 16 + ln];
#pragma unroll
      for (int e = 0; e < 8; e++) {
        const int row = tr * 16 + e + 8 * hf;
        const float v = inn[(size_t)row * kHD + tc * 16 + ln] / csc +
                        acc[tc][e] / isc[e];
        acc[tc][e] = v;
        rsum[e] += v;
        rsq[e] += v * v;
      }
    }
    float mean[8], rstd[8];
#pragma unroll
    for (int e = 0; e < 8; e++) {
      rsum[e] += __shfl_xor(rsum[e], 1);
      rsum[e] += __shfl_xor(rsum[e], 2);
      rsum[e] += __shfl_xor(rsum[e], 4);
      rsum[e] += __shfl_xor(rsum[e], 8);
      rsq[e] += __shfl_xor(rsq[e], 1);
      rsq[e] += __shfl_xor(rsq[e], 2);
      rsq[e] += __shfl_xor(rsq[e], 4);
      rsq[e] += __shfl_xor(rsq[e], 8);
      mean[e] = rsum[e] * (1.f / kHD);
      const float var = rsq[e] * (1.f / kHD) - mean[e] * mean[e];
      rstd[e] = rsqrtf(var + kEPS);
    }

#pragma unroll
    for (int tc = 0; tc < 16; tc++) {
#pragma unroll
      for (int e = 0; e < 8; e++) {
        const int row = tr * 16 + e + 8 * hf;
        const int col = tc * 16 + ln;
        const int tt = n * kC + row;
        const float nv = (acc[tc][e] - mean[e]) * rstd[e];
        const size_t gi = ((size_t)b * kT + tt) * kE2 + h * kHD + col;
        const float gv = (float)gb[gi];
        const float sg = gv / (1.f + __expf(-gv));
        outg[gi] = (bf16_t)(sg * nv);
      }
    }
  }
}

// ---------------------------------------------------------------------------
// host launch
// ---------------------------------------------------------------------------
extern "C" void kernel_launch(void* const* d_in, const int* in_sizes, int n_in,
                              void* d_out, int out_size, void* d_ws,
                              size_t ws_size, hipStream_t stream) {
  (void)in_sizes; (void)n_in; (void)out_size; (void)ws_size;

  const float* x    = (const float*)d_in[0];
  const float* sinp = (const float*)d_in[1];
  const float* cosp = (const float*)d_in[2];
  const float* mask = (const float*)d_in[3];
  const float* cdec = (const float*)d_in[4];
  const float* idec = (const float*)d_in[5];
  const float* Wq = (const float*)d_in[6];   const float* bq = (const float*)d_in[7];
  const float* Wk = (const float*)d_in[8];   const float* bk = (const float*)d_in[9];
  const float* Wv = (const float*)d_in[10];  const float* bv = (const float*)d_in[11];
  const float* Wg = (const float*)d_in[12];  const float* bg = (const float*)d_in[13];
  const float* Wo = (const float*)d_in[14];  const float* bo = (const float*)d_in[15];
  float* out = (float*)d_out;

  char* ws = (char*)d_ws;
  // byte offsets (with two exact-fit aliases)
  const size_t o_xb   = 0;                       // 32 MB  x bf16
  const size_t o_wqb  = o_xb  + 33554432;        // 8 MB
  const size_t o_wkb  = o_wqb + 8388608;         // 8 MB
  const size_t o_wvb  = o_wkb + 8388608;         // 16 MB
  const size_t o_wgb  = o_wvb + 16777216;        // 16 MB
  const size_t o_kvf  = 0;                       // ALIAS: kv f32 (64MB) over xb..wvb (dead after GEMMs)
  const size_t o_wob  = o_wgb + 16777216;        // 16 MB
  const size_t o_qb   = o_wob + 16777216;        // 32 MB
  const size_t o_kb   = o_qb  + 33554432;        // 32 MB
  const size_t o_vb   = o_kb  + 33554432;        // 64 MB
  const size_t o_innerf = o_qb;                  // ALIAS: inner f32 (128MB) over qb..vb (dead after rotary/pack)
  const size_t o_gb   = o_vb  + 67108864;        // 64 MB
  const size_t o_qrb  = o_gb  + 67108864;        // 32 MB
  const size_t o_qdb  = o_qrb + 33554432;        // 32 MB
  const size_t o_krb  = o_qdb + 33554432;        // 32 MB
  const size_t o_krtb = o_krb + 33554432;        // 32 MB
  const size_t o_vtb  = o_krtb + 33554432;       // 64 MB
  const size_t o_vmtb = o_vtb + 67108864;        // 64 MB
  const size_t o_qkb  = o_vmtb + 67108864;       // 64 MB
  const size_t o_isc  = o_qkb + 67108864;        // 0.5 MB
  const size_t o_kvrect = o_isc + 524288;        // 32 MB
  const size_t o_csc  = o_kvrect + 33554432;     // 0.5 MB
  const size_t o_outg = o_csc + 524288;          // 64 MB

  bf16_t* xb   = (bf16_t*)(ws + o_xb);
  bf16_t* wqb  = (bf16_t*)(ws + o_wqb);
  bf16_t* wkb  = (bf16_t*)(ws + o_wkb);
  bf16_t* wvb  = (bf16_t*)(ws + o_wvb);
  bf16_t* wgb  = (bf16_t*)(ws + o_wgb);
  bf16_t* wob  = (bf16_t*)(ws + o_wob);
  bf16_t* qb   = (bf16_t*)(ws + o_qb);
  bf16_t* kb   = (bf16_t*)(ws + o_kb);
  bf16_t* vb   = (bf16_t*)(ws + o_vb);
  bf16_t* gb   = (bf16_t*)(ws + o_gb);
  bf16_t* qrb  = (bf16_t*)(ws + o_qrb);
  bf16_t* qdb  = (bf16_t*)(ws + o_qdb);
  bf16_t* krb  = (bf16_t*)(ws + o_krb);
  bf16_t* krtb = (bf16_t*)(ws + o_krtb);
  bf16_t* vtb  = (bf16_t*)(ws + o_vtb);
  bf16_t* vmtb = (bf16_t*)(ws + o_vmtb);
  bf16_t* qkb  = (bf16_t*)(ws + o_qkb);
  float*  iscale = (float*)(ws + o_isc);
  float*  kvf    = (float*)(ws + o_kvf);
  bf16_t* kvrect = (bf16_t*)(ws + o_kvrect);
  float*  cscale = (float*)(ws + o_csc);
  float*  innerf = (float*)(ws + o_innerf);
  bf16_t* outg   = (bf16_t*)(ws + o_outg);

  const int M = kB * kT;  // 8192

  // K0: convert to bf16
  auto cvt = [&](const float* s, bf16_t* d, int n) {
    msr_f32_to_bf16<<<(n + 255) / 256, 256, 0, stream>>>(s, d, n);
  };
  cvt(x,  xb,  kB * kT * kE);
  cvt(Wq, wqb, kE * kE);
  cvt(Wk, wkb, kE * kE);
  cvt(Wv, wvb, kE2 * kE);
  cvt(Wg, wgb, kE2 * kE);
  cvt(Wo, wob, kE * kE2);

  // K1: projection GEMMs (bf16 out)
  const float kscale = 0.088388347648318447f;  // KD^-0.5
  msr_gemm_bf16<true><<<dim3(kE / 128, M / 128), 256, 0, stream>>>(
      xb, wqb, bq, 1.f, M, kE, kE, nullptr, qb);
  msr_gemm_bf16<true><<<dim3(kE / 128, M / 128), 256, 0, stream>>>(
      xb, wkb, bk, kscale, M, kE, kE, nullptr, kb);
  msr_gemm_bf16<true><<<dim3(kE2 / 128, M / 128), 256, 0, stream>>>(
      xb, wvb, bv, 1.f, M, kE2, kE, nullptr, vb);
  msr_gemm_bf16<true><<<dim3(kE2 / 128, M / 128), 256, 0, stream>>>(
      xb, wgb, bg, 1.f, M, kE2, kE, nullptr, gb);

  // K2: rotary
  msr_rotary<<<(kB * kT * kE / 2 + 255) / 256, 256, 0, stream>>>(
      qb, kb, sinp, cosp, idec, qrb, qdb, krb, krtb);

  // K4: pack v (before qk so the vb region can be re-used by innerf)
  msr_pack_v<<<(kB * kT * kE2 + 255) / 256, 256, 0, stream>>>(vb, mask, vtb, vmtb);

  const int nBNH = kB * kN * kH;  // 512
  // K3: qk + inner_scale
  msr_qk<<<nBNH, 256, 0, stream>>>(qrb, krb, mask, qkb, iscale);
  // K5: inner_output
  msr_inner<<<nBNH, 256, 0, stream>>>(qkb, vtb, innerf);
  // K5b: per-chunk kv
  msr_kv<<<nBNH, 256, 0, stream>>>(krtb, vmtb, kvf);
  // K6: recurrent scan
  msr_scan<<<kB * kH, 256, 0, stream>>>(kvf, cdec, kvrect, cscale);
  // K7: cross + combine + groupnorm + silu gate
  msr_cross<<<nBNH, 256, 0, stream>>>(qdb, kvrect, innerf, cscale, iscale, gb, outg);

  // K8: output projection (f32 out)
  msr_gemm_bf16<false><<<dim3(kE / 128, M / 128), 256, 0, stream>>>(
      outg, wob, bo, 1.f, M, kE, kE2, out, nullptr);
}
